// AdaSparseMDLModel_86955907875137
// MI455X (gfx1250) — compile-verified, matrix operand order checked by reading, not run on hardware
//
#include <hip/hip_runtime.h>

#define B_SZ 16384
#define I_SZ 1280
#define H_SZ 320
#define D_SZ 4
#define N1 512
#define N2 256
#define N3 128

typedef __attribute__((ext_vector_type(16))) __bf16 v16bf;
typedef __attribute__((ext_vector_type(8)))  float  v8f;

union FragU { uint4 q[2]; v16bf v; };

// Load one 16-bit-element WMMA fragment (A or B operand) from a row-major,
// K-contiguous row pointer. Per CDNA5 ISA 16-bit A layout: lane l (0-15)
// holds row M=l; half h = lane>>4 selects K chunks [k0+8h, k0+8h+8) and
// [k0+16+8h, k0+16+8h+8). Two 16-byte loads per lane.
__device__ __forceinline__ v16bf load_frag(const __bf16* rowptr, int k0, int h) {
  FragU u;
  u.q[0] = *(const uint4*)(rowptr + k0 + 8 * h);
  u.q[1] = *(const uint4*)(rowptr + k0 + 16 + 8 * h);
  return u.v;
}

__device__ __forceinline__ v8f wmma_bf16(v16bf a, v16bf b, v8f c) {
  return __builtin_amdgcn_wmma_f32_16x16x32_bf16(false, a, false, b, (short)0, c,
                                                 false, false);
}

__device__ __forceinline__ float sigmoidf_(float x) {
  return 1.0f / (1.0f + __expf(-x));
}

// ---------------- bucketing ----------------
__global__ void zero_cnt_kernel(int* cnt) {
  if (threadIdx.x < D_SZ) cnt[threadIdx.x] = 0;
}

__global__ void fill_perm_kernel(const int* __restrict__ dom, int* cnt,
                                 int* __restrict__ perm) {
  int i = blockIdx.x * blockDim.x + threadIdx.x;
  if (i >= B_SZ) return;
  int d = dom[i];
  d = d < 0 ? 0 : (d > D_SZ - 1 ? D_SZ - 1 : d);
  int slot = atomicAdd(&cnt[d], 1);
  perm[d * B_SZ + slot] = i;
}

// ---------------- weight transpose+convert: src [BATCH][K][N] f32 -> dst [BATCH][N][K] bf16
__global__ void transpose_bf16_kernel(const float* __restrict__ src,
                                      __bf16* __restrict__ dst,
                                      int K, int N, long total) {
  long i = (long)blockIdx.x * blockDim.x + threadIdx.x;
  if (i >= total) return;
  long kn = (long)K * N;
  long b = i / kn;
  long rem = i - b * kn;
  int k = (int)(rem / N);
  int n = (int)(rem - (long)k * N);
  dst[b * kn + (long)n * K + k] = (__bf16)src[i];
}

// ---------------- pruner: per-domain 1280->320 (relu) -> 1280 (sigmoid) -> mask*emb
// 32-row tiles; each wave computes 32x64 output per n-group (8 accumulators).
// Dynamic LDS: As[32][1280] bf16 (80KB) | Hs[32][320] bf16 (20KB) | rows[32]
__global__ __launch_bounds__(128) void pruner_kernel(
    const float* __restrict__ emb, const int* __restrict__ cnt,
    const int* __restrict__ perm,
    const __bf16* __restrict__ w1t,  // [D][H][I]
    const float* __restrict__ b1,    // [D][H]
    const __bf16* __restrict__ w2t,  // [D][I][H]
    const float* __restrict__ b2,    // [D][I]
    __bf16* __restrict__ sparse) {   // [B][I]
  extern __shared__ char smem[];
  __bf16* As = (__bf16*)smem;                       // 32*1280
  __bf16* Hs = As + 32 * I_SZ;                      // 32*320
  int* rows = (int*)(smem + (32 * I_SZ + 32 * H_SZ) * 2);

  int tile = blockIdx.x;
  int d = blockIdx.y;
  int count = cnt[d];
  int m0 = tile * 32;
  if (m0 >= count) return;

  int t = threadIdx.x;
  if (t < 32) {
    int m = m0 + t;
    if (m > count - 1) m = count - 1;
    rows[t] = perm[d * B_SZ + m];
  }
  __syncthreads();

  // gather + convert 32 emb rows into LDS (float4 reads)
  for (int idx = t; idx < 32 * (I_SZ / 4); idx += 128) {
    int mi = idx / (I_SZ / 4);
    int kc = idx - mi * (I_SZ / 4);
    float4 f = *(const float4*)(emb + (size_t)rows[mi] * I_SZ + kc * 4);
    __bf16* p = As + mi * I_SZ + kc * 4;
    p[0] = (__bf16)f.x; p[1] = (__bf16)f.y;
    p[2] = (__bf16)f.z; p[3] = (__bf16)f.w;
  }
  __syncthreads();

  int wave = t >> 5, lane = t & 31, l = lane & 15, h = lane >> 4;

  // GEMM1: h = relu(A @ W1[d] + b1[d]), N = 320 -> 5 groups of 64 cols
  const __bf16* w1base = w1t + (size_t)d * H_SZ * I_SZ;
  for (int g = wave; g < H_SZ / 64; g += 4) {
    int n0 = g * 64;
    v8f acc[8] = {{}, {}, {}, {}, {}, {}, {}, {}};
    const __bf16* arow0 = As + l * I_SZ;
    const __bf16* arow1 = As + (16 + l) * I_SZ;
    const __bf16* brow[4];
#pragma unroll
    for (int i = 0; i < 4; i++) brow[i] = w1base + (size_t)(n0 + 16 * i + l) * I_SZ;
    for (int k0 = 0; k0 < I_SZ; k0 += 32) {
      v16bf a0 = load_frag(arow0, k0, h);
      v16bf a1 = load_frag(arow1, k0, h);
#pragma unroll
      for (int i = 0; i < 4; i++) {
        v16bf b = load_frag(brow[i], k0, h);
        acc[i] = wmma_bf16(a0, b, acc[i]);
        acc[4 + i] = wmma_bf16(a1, b, acc[4 + i]);
      }
    }
#pragma unroll
    for (int i = 0; i < 4; i++) {
      int n = n0 + 16 * i + l;
      float bias = b1[d * H_SZ + n];
#pragma unroll
      for (int s = 0; s < 2; s++) {
#pragma unroll
        for (int j = 0; j < 8; j++) {
          float y = acc[4 * s + i][j] + bias;
          if (y < 0.0f) y = 0.0f;
          Hs[(16 * s + j + 8 * h) * H_SZ + n] = (__bf16)y;
        }
      }
    }
  }
  __syncthreads();

  // GEMM2: p = sigmoid(h @ W2[d] + b2[d]); sparse = emb * p * (p > 0.5)
  // N = 1280 -> 20 groups of 64 cols
  const __bf16* w2base = w2t + (size_t)d * I_SZ * H_SZ;
  int vcount = count - m0;
  if (vcount > 32) vcount = 32;
  for (int g = wave; g < I_SZ / 64; g += 4) {
    int n0 = g * 64;
    v8f acc[8] = {{}, {}, {}, {}, {}, {}, {}, {}};
    const __bf16* arow0 = Hs + l * H_SZ;
    const __bf16* arow1 = Hs + (16 + l) * H_SZ;
    const __bf16* brow[4];
#pragma unroll
    for (int i = 0; i < 4; i++) brow[i] = w2base + (size_t)(n0 + 16 * i + l) * H_SZ;
    for (int k0 = 0; k0 < H_SZ; k0 += 32) {
      v16bf a0 = load_frag(arow0, k0, h);
      v16bf a1 = load_frag(arow1, k0, h);
#pragma unroll
      for (int i = 0; i < 4; i++) {
        v16bf b = load_frag(brow[i], k0, h);
        acc[i] = wmma_bf16(a0, b, acc[i]);
        acc[4 + i] = wmma_bf16(a1, b, acc[4 + i]);
      }
    }
#pragma unroll
    for (int i = 0; i < 4; i++) {
      int n = n0 + 16 * i + l;
      float bias = b2[d * I_SZ + n];
#pragma unroll
      for (int s = 0; s < 2; s++) {
#pragma unroll
        for (int j = 0; j < 8; j++) {
          int mi = 16 * s + j + 8 * h;
          if (mi < vcount) {
            int gr = rows[mi];
            float p = sigmoidf_(acc[4 * s + i][j] + bias);
            float e = emb[(size_t)gr * I_SZ + n];
            float sv = (p > 0.5f) ? e * p : 0.0f;
            sparse[(size_t)gr * I_SZ + n] = (__bf16)sv;
          }
        }
      }
    }
  }
}

// ---------------- generic fused GEMM: out = act(A[B,K] @ Wt[N,K]^T + bias)
// 32-row tiles; each wave computes 32x64 output per n-group (8 accumulators).
// Dynamic LDS: As[32*K] bf16 (<= 80KB)
__global__ __launch_bounds__(128) void mlp_gemm_kernel(
    const __bf16* __restrict__ A, const __bf16* __restrict__ Wt,
    const float* __restrict__ bias, __bf16* __restrict__ out,
    int K, int N, int relu) {
  extern __shared__ char smem[];
  __bf16* As = (__bf16*)smem;
  int t = threadIdx.x;
  size_t row0 = (size_t)blockIdx.x * 32;
  // A rows are contiguous: one coalesced uint4 block copy
  const uint4* srcA = (const uint4*)(A + row0 * K);
  uint4* dstA = (uint4*)As;
  int chunks = 32 * K / 8;
  for (int c = t; c < chunks; c += 128) dstA[c] = srcA[c];
  __syncthreads();

  int wave = t >> 5, lane = t & 31, l = lane & 15, h = lane >> 4;
  for (int g = wave; g < N / 64; g += 4) {
    int n0 = g * 64;
    v8f acc[8] = {{}, {}, {}, {}, {}, {}, {}, {}};
    const __bf16* arow0 = As + l * K;
    const __bf16* arow1 = As + (16 + l) * K;
    const __bf16* brow[4];
#pragma unroll
    for (int i = 0; i < 4; i++) brow[i] = Wt + (size_t)(n0 + 16 * i + l) * K;
    for (int k0 = 0; k0 < K; k0 += 32) {
      v16bf a0 = load_frag(arow0, k0, h);
      v16bf a1 = load_frag(arow1, k0, h);
#pragma unroll
      for (int i = 0; i < 4; i++) {
        v16bf b = load_frag(brow[i], k0, h);
        acc[i] = wmma_bf16(a0, b, acc[i]);
        acc[4 + i] = wmma_bf16(a1, b, acc[4 + i]);
      }
    }
#pragma unroll
    for (int i = 0; i < 4; i++) {
      int n = n0 + 16 * i + l;
      float bi = bias[n];
#pragma unroll
      for (int s = 0; s < 2; s++) {
#pragma unroll
        for (int j = 0; j < 8; j++) {
          float y = acc[4 * s + i][j] + bi;
          if (relu && y < 0.0f) y = 0.0f;
          out[(row0 + 16 * s + j + 8 * h) * N + n] = (__bf16)y;
        }
      }
    }
  }
}

// ---------------- final 128 -> 1 dot + sigmoid
__global__ void final_kernel(const __bf16* __restrict__ A,
                             const float* __restrict__ w4,
                             const float* __restrict__ b4,
                             float* __restrict__ out) {
  int b = blockIdx.x * blockDim.x + threadIdx.x;
  if (b >= B_SZ) return;
  const __bf16* row = A + (size_t)b * N3;
  float s = 0.0f;
#pragma unroll 8
  for (int k = 0; k < N3; k++) s += (float)row[k] * w4[k];
  out[b] = sigmoidf_(s + b4[0]);
}

extern "C" void kernel_launch(void* const* d_in, const int* in_sizes, int n_in,
                              void* d_out, int out_size, void* d_ws, size_t ws_size,
                              hipStream_t stream) {
  const float* emb       = (const float*)d_in[0];
  const int*   domain_id = (const int*)d_in[1];
  const float* p_w1 = (const float*)d_in[2];
  const float* p_b1 = (const float*)d_in[3];
  const float* p_w2 = (const float*)d_in[4];
  const float* p_b2 = (const float*)d_in[5];
  const float* d_w1 = (const float*)d_in[6];
  const float* d_b1 = (const float*)d_in[7];
  const float* d_w2 = (const float*)d_in[8];
  const float* d_b2 = (const float*)d_in[9];
  const float* d_w3 = (const float*)d_in[10];
  const float* d_b3 = (const float*)d_in[11];
  const float* d_w4 = (const float*)d_in[12];
  const float* d_b4 = (const float*)d_in[13];
  float* out = (float*)d_out;

  char* ws = (char*)d_ws;
  size_t off = 0;
  auto alloc = [&](size_t bytes) -> char* {
    char* p = ws + off;
    off = (off + bytes + 255) & ~(size_t)255;
    return p;
  };
  __bf16* w1t    = (__bf16*)alloc((size_t)D_SZ * H_SZ * I_SZ * 2);
  __bf16* w2t    = (__bf16*)alloc((size_t)D_SZ * I_SZ * H_SZ * 2);
  __bf16* dw1t   = (__bf16*)alloc((size_t)N1 * I_SZ * 2);
  __bf16* dw2t   = (__bf16*)alloc((size_t)N2 * N1 * 2);
  __bf16* dw3t   = (__bf16*)alloc((size_t)N3 * N2 * 2);
  __bf16* sparse = (__bf16*)alloc((size_t)B_SZ * I_SZ * 2);
  __bf16* a1     = (__bf16*)alloc((size_t)B_SZ * N1 * 2);
  __bf16* a2     = (__bf16*)alloc((size_t)B_SZ * N2 * 2);
  __bf16* a3     = (__bf16*)alloc((size_t)B_SZ * N3 * 2);
  int*    perm   = (int*)alloc((size_t)D_SZ * B_SZ * 4);
  int*    cnt    = (int*)alloc(64);

  // bucket rows by domain
  zero_cnt_kernel<<<1, 32, 0, stream>>>(cnt);
  fill_perm_kernel<<<B_SZ / 256, 256, 0, stream>>>(domain_id, cnt, perm);

  // weight convert + transpose (K-contiguous bf16 for B-operand loads)
  {
    long tot;
    tot = (long)D_SZ * I_SZ * H_SZ;
    transpose_bf16_kernel<<<(tot + 255) / 256, 256, 0, stream>>>(p_w1, w1t, I_SZ, H_SZ, tot);
    tot = (long)D_SZ * H_SZ * I_SZ;
    transpose_bf16_kernel<<<(tot + 255) / 256, 256, 0, stream>>>(p_w2, w2t, H_SZ, I_SZ, tot);
    tot = (long)I_SZ * N1;
    transpose_bf16_kernel<<<(tot + 255) / 256, 256, 0, stream>>>(d_w1, dw1t, I_SZ, N1, tot);
    tot = (long)N1 * N2;
    transpose_bf16_kernel<<<(tot + 255) / 256, 256, 0, stream>>>(d_w2, dw2t, N1, N2, tot);
    tot = (long)N2 * N3;
    transpose_bf16_kernel<<<(tot + 255) / 256, 256, 0, stream>>>(d_w3, dw3t, N2, N3, tot);
  }

  // domain-routed pruner (grid.y = domain; early-exit past bucket count)
  size_t pruner_lds = (size_t)(32 * I_SZ + 32 * H_SZ) * 2 + 32 * 4;
  pruner_kernel<<<dim3(B_SZ / 32, D_SZ), 128, pruner_lds, stream>>>(
      emb, cnt, perm, w1t, p_b1, w2t, p_b2, sparse);

  // shared MLP (dynamic LDS = 32*K bf16)
  mlp_gemm_kernel<<<B_SZ / 32, 128, (size_t)32 * I_SZ * 2, stream>>>(
      sparse, dw1t, d_b1, a1, I_SZ, N1, 1);
  mlp_gemm_kernel<<<B_SZ / 32, 128, (size_t)32 * N1 * 2, stream>>>(
      a1, dw2t, d_b2, a2, N1, N2, 1);
  mlp_gemm_kernel<<<B_SZ / 32, 128, (size_t)32 * N2 * 2, stream>>>(
      a2, dw3t, d_b3, a3, N2, N3, 1);

  // 128 -> 1 + sigmoid
  final_kernel<<<B_SZ / 256, 256, 0, stream>>>(a3, d_w4, d_b4, out);
}
